// HeteroGraphormer_17489106829951
// MI455X (gfx1250) — matrix operand order repaired and checked by reading, not compiled, not run on hardware
//
#include <hip/hip_runtime.h>
#include <hip/hip_bf16.h>
#include <math.h>

typedef __bf16 bf16;
typedef __attribute__((ext_vector_type(8)))  bf16  v8bf;
typedef __attribute__((ext_vector_type(16))) bf16  v16bf;
typedef __attribute__((ext_vector_type(8)))  float v8f;

#define NN 2048
#define CC 512
#define HH 8
#define DD 64
#define FF 2048
#define LL 3

// ---------- helpers ----------
__device__ __forceinline__ bf16 tobf(float f) {
  union { float f; unsigned u; } c; c.f = f;
  unsigned r = (c.u + 0x7FFFu + ((c.u >> 16) & 1u)) >> 16;
  unsigned short s = (unsigned short)r;
  bf16 o; __builtin_memcpy(&o, &s, 2); return o;
}

// CDNA5 async DMA: global -> LDS, 16 bytes, tracked by ASYNCcnt
__device__ __forceinline__ void async_b128(const bf16* gsrc, void* ldst) {
  unsigned loff = (unsigned)(uintptr_t)ldst;               // low 32 bits = LDS offset
  unsigned long long ga = (unsigned long long)(uintptr_t)gsrc;
  asm volatile("global_load_async_to_lds_b128 %0, %1, off"
               :: "v"(loff), "v"(ga) : "memory");
}
__device__ __forceinline__ void wait_async0() {
  asm volatile("s_wait_asynccnt 0x0" ::: "memory");
}

// ---------- f32 -> bf16 transpose convert: x[R][Cn] -> y[Cn][R] ----------
__global__ __launch_bounds__(256) void convert_transpose(const float* __restrict__ x,
                                                         bf16* __restrict__ y,
                                                         int R, int Cn) {
  size_t lbase = (size_t)blockIdx.y * R * Cn;
  int i = blockIdx.x * 256 + threadIdx.x;
  if (i < R * Cn) {
    int r = i / Cn, c = i - r * Cn;
    y[lbase + (size_t)c * R + r] = tobf(x[lbase + i]);
  }
}

// ---------- input embedding ----------
__global__ __launch_bounds__(256) void embed_kernel(
    const float* __restrict__ feats, const int* __restrict__ ind,
    const int* __restrict__ outd, const float* __restrict__ ide,
    const float* __restrict__ ode, float* __restrict__ xf, bf16* __restrict__ xb) {
  int i = blockIdx.x * 256 + threadIdx.x;      // N*C elements
  int n = i / CC, c = i - n * CC;
  int a = min(max(ind[n], 0), 31);
  int b = min(max(outd[n], 0), 31);
  float v = feats[i] + ide[a * CC + c] + ode[b * CC + c];
  xf[i] = v; xb[i] = tobf(v);
}

// ---------- WMMA bf16 GEMM:  C = A @ B^T (+bias)(+epi) ----------
// A: [M][K] row-major (lda). B: [Nd][K] row-major (ldb)  -> C[m][n]=sum_k A[m][k]*B[n][k]
// epi: 0 = bias add, 1 = bias + exact gelu, 2 = score (scale + spatial bias)
// ctrans: store Cb transposed ([n][m])
template <int BN>
__global__ __launch_bounds__(256) void wmma_gemm(
    const bf16* __restrict__ A, int lda,
    const bf16* __restrict__ B, int ldb,
    const float* __restrict__ bias,
    float* __restrict__ Cf, int ldc,
    bf16* __restrict__ Cb, int ldcb, int ctrans,
    int K, int epi, float scale,
    const int* __restrict__ dist, int lddist,
    const float* __restrict__ sp, int head) {
  constexpr int BM = 128, BK = 32, LDS_S = BK + 8;
  constexpr int WN = BN / 2;       // per-wave N extent (8 waves as 4x2)
  constexpr int NF = WN / 16;      // N fragments per wave
  __shared__ bf16 Asb[2][BM][LDS_S];
  __shared__ bf16 Bsb[2][BN][LDS_S];

  const int tid  = threadIdx.x;
  const int lane = tid & 31;
  const int wave = tid >> 5;
  const int wm   = wave & 3;       // 4 waves along M
  const int wn   = wave >> 2;      // 2 waves along N
  const int bm0  = blockIdx.y * BM;
  const int bn0  = blockIdx.x * BN;

  // fixed per-thread staging assignment: chunk = 16B (8 bf16) of one tile row
  const int ra  = tid >> 2;            // A rows: ra and ra+64
  const int kca = (tid & 3) * 8;       // k-offset within tile
  const bf16* gA0 = A + (size_t)(bm0 + ra)      * lda + kca;
  const bf16* gA1 = A + (size_t)(bm0 + ra + 64) * lda + kca;
  const bf16* gB0 = B + (size_t)(bn0 + (ra % BN))      * ldb + kca;
  const bf16* gB1 = B + (size_t)(bn0 + (ra % BN) + 64) * ldb + kca;  // only BN==128

  auto issue = [&](int buf, int koff) {
    async_b128(gA0 + koff, &Asb[buf][ra][kca]);
    async_b128(gA1 + koff, &Asb[buf][ra + 64][kca]);
    async_b128(gB0 + koff, &Bsb[buf][ra % BN][kca]);
    if (BN == 128) async_b128(gB1 + koff, &Bsb[buf][ra + 64][kca]);
  };

  v8f acc[2][NF] = {};
  const int nk = K / BK;

  issue(0, 0);
  for (int it = 0; it < nk; ++it) {
    const int cur = it & 1;
    wait_async0();                 // my async DMAs for tile `cur` done
    __syncthreads();               // everyone's DMAs done; prev compute on other buf done
    if (it + 1 < nk) issue(cur ^ 1, (it + 1) * BK);

    // --- fragments per ISA 16-bit layouts ---
    const int lrow = lane & 15;
    const int koff = (lane >> 4) * 8;    // A: lanes16-31 start at K=8 (+16 second chunk)
    const int kbh  = (lane >> 4) * 8;    // B: lanes16-31 start at K=8 (+16 second chunk)
    v16bf afrag[2];
#pragma unroll
    for (int mi = 0; mi < 2; ++mi) {
      const bf16* p = &Asb[cur][wm * 32 + mi * 16 + lrow][koff];
      v8bf lo = *(const v8bf*)p;
      v8bf hi = *(const v8bf*)(p + 16);
      afrag[mi] = __builtin_shufflevector(lo, hi, 0,1,2,3,4,5,6,7,8,9,10,11,12,13,14,15);
    }
#pragma unroll
    for (int ni = 0; ni < NF; ++ni) {
      const bf16* p = &Bsb[cur][wn * WN + ni * 16 + lrow][kbh];
      v8bf lo = *(const v8bf*)p;
      v8bf hi = *(const v8bf*)(p + 16);
      v16bf bfrag = __builtin_shufflevector(lo, hi, 0,1,2,3,4,5,6,7,8,9,10,11,12,13,14,15);
#pragma unroll
      for (int mi = 0; mi < 2; ++mi)
        acc[mi][ni] = __builtin_amdgcn_wmma_f32_16x16x32_bf16(
            false, afrag[mi], false, bfrag, (short)0, acc[mi][ni], false, false);
    }
  }

  // --- epilogue: C/D layout: VGPR r -> M = r + 8*(lane>=16), N = lane&15 ---
  const int lcol = lane & 15;
  const int rsel = (lane >> 4) * 8;
#pragma unroll
  for (int mi = 0; mi < 2; ++mi) {
#pragma unroll
    for (int ni = 0; ni < NF; ++ni) {
      int gn  = bn0 + wn * WN + ni * 16 + lcol;
      int gmb = bm0 + wm * 32 + mi * 16 + rsel;
#pragma unroll
      for (int r = 0; r < 8; ++r) {
        int gm  = gmb + r;
        float v = acc[mi][ni][r];
        if (epi == 2) {
          v = v * scale + sp[(size_t)dist[(size_t)gm * lddist + gn] * HH + head];
        } else {
          if (bias) v += bias[gn];
          if (epi == 1) v = 0.5f * v * (1.0f + erff(v * 0.70710678118654752f));
        }
        if (Cf) Cf[(size_t)gm * ldc + gn] = v;
        if (Cb) {
          if (ctrans) Cb[(size_t)gn * ldcb + gm] = tobf(v);
          else        Cb[(size_t)gm * ldcb + gn] = tobf(v);
        }
      }
    }
  }
}

// ---------- row softmax: f32 scores -> bf16 probs ----------
__global__ __launch_bounds__(256) void softmax_rows(const float* __restrict__ S,
                                                    bf16* __restrict__ P) {
  __shared__ float red[256];
  int row = blockIdx.x, tid = threadIdx.x;
  const float* s = S + (size_t)row * NN;
  float vals[NN / 256];
  float vmax = -3.0e38f;
#pragma unroll
  for (int i = 0; i < NN / 256; ++i) { vals[i] = s[tid + i * 256]; vmax = fmaxf(vmax, vals[i]); }
  red[tid] = vmax; __syncthreads();
  for (int st = 128; st > 0; st >>= 1) { if (tid < st) red[tid] = fmaxf(red[tid], red[tid + st]); __syncthreads(); }
  vmax = red[0]; __syncthreads();
  float sum = 0.f;
#pragma unroll
  for (int i = 0; i < NN / 256; ++i) { vals[i] = __expf(vals[i] - vmax); sum += vals[i]; }
  red[tid] = sum; __syncthreads();
  for (int st = 128; st > 0; st >>= 1) { if (tid < st) red[tid] += red[tid + st]; __syncthreads(); }
  float inv = 1.0f / red[0];
  bf16* p = P + (size_t)row * NN;
#pragma unroll
  for (int i = 0; i < NN / 256; ++i) p[tid + i * 256] = tobf(vals[i] * inv);
}

// ---------- residual + layernorm ----------
__global__ __launch_bounds__(256) void residual_ln(
    const float* __restrict__ x, const float* __restrict__ d,
    const float* __restrict__ g, const float* __restrict__ b,
    float* __restrict__ outf, bf16* __restrict__ outb) {
  __shared__ float red[256];
  int row = blockIdx.x, tid = threadIdx.x;
  size_t base = (size_t)row * CC;
  float v0 = x[base + tid] + d[base + tid];
  float v1 = x[base + tid + 256] + d[base + tid + 256];
  red[tid] = v0 + v1; __syncthreads();
  for (int st = 128; st > 0; st >>= 1) { if (tid < st) red[tid] += red[tid + st]; __syncthreads(); }
  float mu = red[0] * (1.0f / CC); __syncthreads();
  float d0 = v0 - mu, d1 = v1 - mu;
  red[tid] = d0 * d0 + d1 * d1; __syncthreads();
  for (int st = 128; st > 0; st >>= 1) { if (tid < st) red[tid] += red[tid + st]; __syncthreads(); }
  float rinv = rsqrtf(red[0] * (1.0f / CC) + 1e-5f);
  float y0 = d0 * rinv * g[tid] + b[tid];
  float y1 = d1 * rinv * g[tid + 256] + b[tid + 256];
  outf[base + tid] = y0; outf[base + tid + 256] = y1;
  if (outb) { outb[base + tid] = tobf(y0); outb[base + tid + 256] = tobf(y1); }
}

// ---------- host side ----------
extern "C" void kernel_launch(void* const* d_in, const int* in_sizes, int n_in,
                              void* d_out, int out_size, void* d_ws, size_t ws_size,
                              hipStream_t stream) {
  const float* feats = (const float*)d_in[0];
  const int*   dist  = (const int*)d_in[1];
  const int*   indeg = (const int*)d_in[2];
  const int*   outdeg= (const int*)d_in[3];
  const float* spemb = (const float*)d_in[4];
  const float* idemb = (const float*)d_in[5];
  const float* odemb = (const float*)d_in[6];
  const float* Wq = (const float*)d_in[7];  const float* bq = (const float*)d_in[8];
  const float* Wk = (const float*)d_in[9];  const float* bk = (const float*)d_in[10];
  const float* Wv = (const float*)d_in[11]; const float* bv = (const float*)d_in[12];
  const float* Wo = (const float*)d_in[13]; const float* bo = (const float*)d_in[14];
  const float* ln1g = (const float*)d_in[15]; const float* ln1b = (const float*)d_in[16];
  const float* ln2g = (const float*)d_in[17]; const float* ln2b = (const float*)d_in[18];
  const float* W1 = (const float*)d_in[19]; const float* b1 = (const float*)d_in[20];
  const float* W2 = (const float*)d_in[21]; const float* b2 = (const float*)d_in[22];

  // workspace carve-up (256B aligned slices)
  char* p = (char*)d_ws;
  auto alloc = [&](size_t bytes) -> void* {
    void* r = (void*)p; p += (bytes + 255) & ~(size_t)255; return r;
  };
  float* xf     = (float*)alloc((size_t)NN * CC * 4);
  float* proj   = (float*)alloc((size_t)NN * CC * 4);
  bf16*  xb     = (bf16*) alloc((size_t)NN * CC * 2);
  bf16*  qb     = (bf16*) alloc((size_t)NN * CC * 2);
  bf16*  kb     = (bf16*) alloc((size_t)NN * CC * 2);
  bf16*  vbT    = (bf16*) alloc((size_t)CC * NN * 2);   // V stored transposed [C][N]
  bf16*  attnb  = (bf16*) alloc((size_t)NN * CC * 2);
  float* scores = (float*)alloc((size_t)NN * NN * 4);
  bf16*  probs  = (bf16*) alloc((size_t)NN * NN * 2);
  bf16*  hb     = (bf16*) alloc((size_t)NN * FF * 2);
  bf16*  wqT    = (bf16*) alloc((size_t)LL * CC * CC * 2);  // [out][in]
  bf16*  wkT    = (bf16*) alloc((size_t)LL * CC * CC * 2);
  bf16*  wvT    = (bf16*) alloc((size_t)LL * CC * CC * 2);
  bf16*  woT    = (bf16*) alloc((size_t)LL * CC * CC * 2);
  bf16*  w1T    = (bf16*) alloc((size_t)LL * FF * CC * 2);  // [F][C]
  bf16*  w2T    = (bf16*) alloc((size_t)LL * CC * FF * 2);  // [C][F]

  // weight conversion + transpose (every call; no cached state)
  {
    dim3 g1((CC * CC + 255) / 256, LL), g2((CC * FF + 255) / 256, LL);
    convert_transpose<<<g1, 256, 0, stream>>>(Wq, wqT, CC, CC);
    convert_transpose<<<g1, 256, 0, stream>>>(Wk, wkT, CC, CC);
    convert_transpose<<<g1, 256, 0, stream>>>(Wv, wvT, CC, CC);
    convert_transpose<<<g1, 256, 0, stream>>>(Wo, woT, CC, CC);
    convert_transpose<<<g2, 256, 0, stream>>>(W1, w1T, CC, FF);  // [C][F] -> [F][C]
    convert_transpose<<<g2, 256, 0, stream>>>(W2, w2T, FF, CC);  // [F][C] -> [C][F]
  }

  // x = feats + in_deg_emb + out_deg_emb
  embed_kernel<<<(NN * CC) / 256, 256, 0, stream>>>(feats, indeg, outdeg, idemb, odemb, xf, xb);

  const float iscale = 0.125f;  // 1/sqrt(64)
  for (int l = 0; l < LL; ++l) {
    const bf16* wq_l = wqT + (size_t)l * CC * CC;
    const bf16* wk_l = wkT + (size_t)l * CC * CC;
    const bf16* wv_l = wvT + (size_t)l * CC * CC;
    const bf16* wo_l = woT + (size_t)l * CC * CC;
    const bf16* w1_l = w1T + (size_t)l * CC * FF;
    const bf16* w2_l = w2T + (size_t)l * FF * CC;

    dim3 gC(CC / 128, NN / 128);   // [N,C] outputs
    dim3 gF(FF / 128, NN / 128);   // [N,F] outputs
    dim3 gS(NN / 128, NN / 128);   // [N,N] scores
    dim3 gO(1, NN / 128);          // [N,64] per-head out

    // Q, K projections (bf16, row-major); V projection stored transposed
    wmma_gemm<128><<<gC, 256, 0, stream>>>(xb, CC, wq_l, CC, bq + l * CC,
        (float*)nullptr, 0, qb, CC, 0, CC, 0, 0.f, nullptr, 0, nullptr, 0);
    wmma_gemm<128><<<gC, 256, 0, stream>>>(xb, CC, wk_l, CC, bk + l * CC,
        (float*)nullptr, 0, kb, CC, 0, CC, 0, 0.f, nullptr, 0, nullptr, 0);
    wmma_gemm<128><<<gC, 256, 0, stream>>>(xb, CC, wv_l, CC, bv + l * CC,
        (float*)nullptr, 0, vbT, NN, 1 /*ctrans*/, CC, 0, 0.f, nullptr, 0, nullptr, 0);

    // attention per head
    for (int h = 0; h < HH; ++h) {
      // scores = q_h @ k_h^T * scale + spatial_emb[dist][h]
      wmma_gemm<128><<<gS, 256, 0, stream>>>(qb + h * DD, CC, kb + h * DD, CC,
          (const float*)nullptr, scores, NN, (bf16*)nullptr, 0, 0,
          DD, 2, iscale, dist, NN, spemb, h);
      softmax_rows<<<NN, 256, 0, stream>>>(scores, probs);
      // out_h = probs @ v_h : B = vbT[h*64 .. h*64+63][:]  (already [d][m] layout)
      wmma_gemm<64><<<gO, 256, 0, stream>>>(probs, NN, vbT + (size_t)(h * DD) * NN, NN,
          (const float*)nullptr, (float*)nullptr, 0, attnb + h * DD, CC, 0,
          NN, 0, 0.f, nullptr, 0, nullptr, 0);
    }

    // output projection -> proj (f32)
    wmma_gemm<128><<<gC, 256, 0, stream>>>(attnb, CC, wo_l, CC, bo + l * CC,
        proj, CC, (bf16*)nullptr, 0, 0, CC, 0, 0.f, nullptr, 0, nullptr, 0);
    // x = LN1(x + proj)
    residual_ln<<<NN, 256, 0, stream>>>(xf, proj, ln1g + l * CC, ln1b + l * CC, xf, xb);

    // FFN: h = gelu(x @ W1 + b1) -> hb (bf16); proj = h @ W2 + b2 (f32)
    wmma_gemm<128><<<gF, 256, 0, stream>>>(xb, CC, w1_l, CC, b1 + l * FF,
        (float*)nullptr, 0, hb, FF, 0, CC, 1, 0.f, nullptr, 0, nullptr, 0);
    wmma_gemm<128><<<gC, 256, 0, stream>>>(hb, FF, w2_l, FF, b2 + l * CC,
        proj, CC, (bf16*)nullptr, 0, 0, FF, 0, 0.f, nullptr, 0, nullptr, 0);

    // x = LN2(x + proj); last layer writes straight into d_out (f32)
    float* outf = (l == LL - 1) ? (float*)d_out : xf;
    bf16*  outb = (l == LL - 1) ? (bf16*)nullptr : xb;
    residual_ln<<<NN, 256, 0, stream>>>(xf, proj, ln2g + l * CC, ln2b + l * CC, outf, outb);
  }
}
